// LEnsemble_62294205662025
// MI455X (gfx1250) — matrix operand order repaired and checked by reading, not compiled, not run on hardware
//
#include <hip/hip_runtime.h>
#include <hip/hip_bf16.h>
#include <math.h>

typedef __bf16 bf16_t;
typedef __attribute__((ext_vector_type(4)))  __bf16 v4bf;
typedef __attribute__((ext_vector_type(8)))  __bf16 v8bf;
typedef __attribute__((ext_vector_type(16))) __bf16 v16bf;
typedef __attribute__((ext_vector_type(8)))  float  v8f;

#define NDIM     256   // ground-set size n
#define KDIM      64   // feature rank k
#define BATCHN  2048
#define MTSTR    264   // halves per row of Mt (256 + pad, 528B = 16B multiple)
#define SGSTR     65   // f32 row stride for Gram G (pad)
#define NTHREADS 128   // 4 wave32s

union frag_u { v16bf v; v8bf h[2]; };

// logdet( M^T M (+ I) ), M = diag(mask)*B, via WMMA bf16 Gram (Mt staged
// transposed in LDS so fragments load as ds_load_b128) + in-LDS Cholesky.
__device__ __forceinline__ float gram_cholesky_logdet(
    const float* __restrict__ B,       // [NDIM][KDIM] row-major f32
    const int*  __restrict__ xrow,     // [NDIM] 0/1 mask, or nullptr (=ones)
    bool add_identity,
    bf16_t* sMt, float* sG, float* sRed, float* sXf)
{
    const int tid = threadIdx.x;

    // ---- stage mask as float ----
    for (int n = tid; n < NDIM; n += NTHREADS)
        sXf[n] = xrow ? (float)xrow[n] : 1.0f;
    __syncthreads();

    // ---- stage Mt[k][n] = x[n]*B[n][k], 4 bf16 per ds_store_b64 ----
    for (int t = tid; t < KDIM * (NDIM / 4); t += NTHREADS) {
        const int k  = t & 63;          // consecutive tid -> consecutive k
        const int n0 = (t >> 6) << 2;
        v4bf p;
#pragma unroll
        for (int i = 0; i < 4; ++i)
            p[i] = (bf16_t)(B[(n0 + i) * KDIM + k] * sXf[n0 + i]);
        *(v4bf*)(sMt + k * MTSTR + n0) = p;
    }
    __syncthreads();

    // ---- G = M^T M : 4 waves x 4 col-tiles, K swept 32/chunk, unrolled ----
    const int lane  = tid & 31;
    const int wave  = tid >> 5;
    const int col   = lane & 15;
    const int half  = lane >> 4;
    const int iBase = wave * 16;

    v8f acc0 = {}, acc1 = {}, acc2 = {}, acc3 = {};
    const bf16_t* rowA = sMt + (iBase + col) * MTSTR;

#pragma unroll
    for (int kk = 0; kk < NDIM; kk += 32) {
        // A fragment 16x32 bf16: two contiguous 16B runs per lane.
        frag_u a;
        a.h[0] = *(const v8bf*)(rowA + kk + half * 16);       // e0..7 : k=half*16+e
        a.h[1] = *(const v8bf*)(rowA + kk + half * 16 + 8);   // e8..15: k=half*16+e
        // B fragments 32x16 bf16: 16 contiguous halves per lane.
        frag_u b0, b1, b2, b3;
        {
            const bf16_t* r0 = sMt + (0 * 16 + col) * MTSTR + kk + half * 16;
            const bf16_t* r1 = sMt + (1 * 16 + col) * MTSTR + kk + half * 16;
            const bf16_t* r2 = sMt + (2 * 16 + col) * MTSTR + kk + half * 16;
            const bf16_t* r3 = sMt + (3 * 16 + col) * MTSTR + kk + half * 16;
            b0.h[0] = *(const v8bf*)(r0); b0.h[1] = *(const v8bf*)(r0 + 8);
            b1.h[0] = *(const v8bf*)(r1); b1.h[1] = *(const v8bf*)(r1 + 8);
            b2.h[0] = *(const v8bf*)(r2); b2.h[1] = *(const v8bf*)(r2 + 8);
            b3.h[0] = *(const v8bf*)(r3); b3.h[1] = *(const v8bf*)(r3 + 8);
        }
        acc0 = __builtin_amdgcn_wmma_f32_16x16x32_bf16(false, a.v, false, b0.v, (short)0, acc0, false, false);
        acc1 = __builtin_amdgcn_wmma_f32_16x16x32_bf16(false, a.v, false, b1.v, (short)0, acc1, false, false);
        acc2 = __builtin_amdgcn_wmma_f32_16x16x32_bf16(false, a.v, false, b2.v, (short)0, acc2, false, false);
        acc3 = __builtin_amdgcn_wmma_f32_16x16x32_bf16(false, a.v, false, b3.v, (short)0, acc3, false, false);
    }

    // ---- spill G tiles (C/D layout: VGPR r -> row r + 8*half) ----
#pragma unroll
    for (int r = 0; r < 8; ++r) {
        const int i = iBase + r + 8 * half;
        sG[i * SGSTR + col +  0] = acc0[r];
        sG[i * SGSTR + col + 16] = acc1[r];
        sG[i * SGSTR + col + 32] = acc2[r];
        sG[i * SGSTR + col + 48] = acc3[r];
    }
    __syncthreads();

    if (add_identity) {
        for (int d = tid; d < KDIM; d += NTHREADS) sG[d * SGSTR + d] += 1.0f;
        __syncthreads();
    }

    // ---- right-looking Cholesky on 64x64 G in LDS (shift/mask indexing) ----
    for (int k = 0; k < KDIM; ++k) {
        if (tid == 0) sG[k * SGSTR + k] = sqrtf(sG[k * SGSTR + k]);
        __syncthreads();
        const float dinv = 1.0f / sG[k * SGSTR + k];
        for (int i = k + 1 + tid; i < KDIM; i += NTHREADS)
            sG[i * SGSTR + k] *= dinv;
        __syncthreads();
        const int rows = KDIM - 1 - k;
        for (int idx = tid; idx < rows * KDIM; idx += NTHREADS) {
            const int i = k + 1 + (idx >> 6);
            const int j = idx & 63;
            if (j > k)
                sG[i * SGSTR + j] -= sG[i * SGSTR + k] * sG[j * SGSTR + k];
        }
        __syncthreads();
    }

    // ---- logdet = 2 * sum(log(diag)) ----
    float part = 0.0f;
    for (int d = tid; d < KDIM; d += NTHREADS)
        part += logf(sG[d * SGSTR + d]);
    sRed[tid] = part;
    __syncthreads();
    for (int s = NTHREADS / 2; s > 0; s >>= 1) {
        if (tid < s) sRed[tid] += sRed[tid + s];
        __syncthreads();
    }
    return 2.0f * sRed[0];
}

// Normalizer: logdet(L0 + I_256) == logdet(I_64 + B^T B)  (Weinstein-Aronszajn)
__global__ void __launch_bounds__(NTHREADS)
lensemble_norm_kernel(const float* __restrict__ B, float* __restrict__ ws) {
    __shared__ bf16_t sMt[KDIM * MTSTR];
    __shared__ float  sG[KDIM * SGSTR];
    __shared__ float  sRed[NTHREADS];
    __shared__ float  sXf[NDIM];
    const float ld0 = gram_cholesky_logdet(B, nullptr, true, sMt, sG, sRed, sXf);
    if (threadIdx.x == 0) ws[0] = ld0;
}

// Per-batch: logdet(G_b) - logdet0, one block per batch element.
__global__ void __launch_bounds__(NTHREADS)
lensemble_batch_kernel(const int* __restrict__ x, const float* __restrict__ B,
                       const float* __restrict__ ws, float* __restrict__ out) {
    __shared__ bf16_t sMt[KDIM * MTSTR];
    __shared__ float  sG[KDIM * SGSTR];
    __shared__ float  sRed[NTHREADS];
    __shared__ float  sXf[NDIM];
    const int b = blockIdx.x;
    const float ld = gram_cholesky_logdet(B, x + (size_t)b * NDIM, false, sMt, sG, sRed, sXf);
    if (threadIdx.x == 0) out[b] = ld - ws[0];
}

extern "C" void kernel_launch(void* const* d_in, const int* in_sizes, int n_in,
                              void* d_out, int out_size, void* d_ws, size_t ws_size,
                              hipStream_t stream) {
    const int*   x = (const int*)d_in[0];    // [2048, 256] {0,1}
    const float* B = (const float*)d_in[1];  // [256, 64] f32
    float* out = (float*)d_out;              // [2048] f32
    float* ws  = (float*)d_ws;               // ws[0] = logdet(L0 + I)

    lensemble_norm_kernel<<<1, NTHREADS, 0, stream>>>(B, ws);
    lensemble_batch_kernel<<<BATCHN, NTHREADS, 0, stream>>>(x, B, ws, out);
    (void)in_sizes; (void)n_in; (void)out_size; (void)ws_size;
}
